// SSP_6828998001545
// MI455X (gfx1250) — matrix operand order repaired and checked by reading, not compiled, not run on hardware
//
#include <hip/hip_runtime.h>
#include <hip/hip_bf16.h>
#include <math.h>

typedef __attribute__((ext_vector_type(16))) _Float16 v16h;
typedef __attribute__((ext_vector_type(8)))  float    v8f;

#define N_NODES   100000
#define E_EDGES   1600000
#define D_IN      512
#define D_HID     256
#define D_OUT     40
#define D_OUT_PAD 48
#define MT        (N_NODES/16)     // 6250 row tiles (exact)
#define KS1       (D_IN/32)        // 16 k-steps, layer 1
#define KS2       (D_HID/32)       // 8 k-steps, layer 2
#define NT1       (D_HID/16)       // 16 col tiles, layer 1
#define NT2       (D_OUT_PAD/16)   // 3 col tiles, layer 2

// ---------------------------------------------------------------------------
// A-fragment element mapping (16-bit A 16x32, CDNA5 ISA 7.12.2):
// lane L: M = L%16, half = L/16. element j: v=j>>1, sub=j&1,
//   k_local = (v<4 ? 2v : 16+2(v-4)) + sub + half*8
// B-fragment (16-bit B 32x16): lane L: N = L%16, half = L/16, k_local = (j&8)*2 + half*8 + (j&7) (same k order as A)
// Fragment blocks stored contiguously: 32 lanes x 16 halves = 1024 B / block.
// ---------------------------------------------------------------------------

__device__ __forceinline__ int a_klocal(int j, int half) {
    int v = j >> 1, sub = j & 1;
    return (v < 4 ? 2 * v : 16 + 2 * (v - 4)) + sub + half * 8;
}

// CDNA5 async copy global->LDS (ASYNCcnt path, CDNA5 ISA).
// loff: LDS byte address (VDST vgpr), gaddr: 64-bit global address (VADDR pair).
// IMPORTANT: loff must be derived from the __shared__ array's pointer so the
// array's address escapes into the asm -- otherwise GlobalsModRef proves the
// memory-clobbering asm can't write the (internal, never-stored) LDS global
// and the subsequent ds loads fold to undef (observed in round 2).
__device__ __forceinline__ void async_g2l_b128(unsigned loff, const void* gaddr) {
    asm volatile("global_load_async_to_lds_b128 %0, %1, off"
                 :: "v"(loff), "v"((unsigned long long)gaddr) : "memory");
}
__device__ __forceinline__ void wait_asynccnt0() {
    asm volatile("s_wait_asynccnt 0x0" ::: "memory");
}

// ---------------- zero fill (graph-capture-safe, no memset API) -------------
__global__ void k_zero_f4(float4* p, int n4) {
    int t = blockIdx.x * blockDim.x + threadIdx.x;
    if (t < n4) p[t] = make_float4(0.f, 0.f, 0.f, 0.f);
}

// ---------------- degree / normalization ------------------------------------
__global__ void k_deg_init(float* deg) {
    int t = blockIdx.x * blockDim.x + threadIdx.x;
    if (t < N_NODES) deg[t] = 1.0f;                 // self-loop
}
__global__ void k_deg_count(const int* dst, float* deg) {
    int t = blockIdx.x * blockDim.x + threadIdx.x;
    if (t < E_EDGES) atomicAdd(&deg[dst[t]], 1.0f);
}
__global__ void k_dinv(float* deg) {
    int t = blockIdx.x * blockDim.x + threadIdx.x;
    if (t < N_NODES) deg[t] = rsqrtf(deg[t]);       // deg >= 1 always
}

// ---------------- fragment-swizzle conversion kernels -----------------------
// x (f32 row-major [N,512]) -> xswz (f16 A-fragment blocks, KS1 ksteps)
__global__ void k_swz_x(const float* __restrict__ x, _Float16* __restrict__ xswz) {
    int t = blockIdx.x * blockDim.x + threadIdx.x;   // over N*512 elements
    if (t >= N_NODES * D_IN) return;
    int j     = t & 15;
    int lane  = (t >> 4) & 31;
    int ksm   = t >> 9;
    int kstep = ksm & (KS1 - 1);
    int mtile = ksm >> 4;                            // / KS1
    int row   = mtile * 16 + (lane & 15);
    int half  = lane >> 4;
    int K     = kstep * 32 + a_klocal(j, half);
    xswz[t] = (_Float16)x[(size_t)row * D_IN + K];
}

// W1 (f32 [512,256]) -> B-fragment blocks, block id = kstep*NT1 + ntile
__global__ void k_swz_w1(const float* __restrict__ W1, _Float16* __restrict__ w1swz) {
    int t = blockIdx.x * blockDim.x + threadIdx.x;   // 512*256 halves
    if (t >= D_IN * D_HID) return;
    int j     = t & 15;
    int lane  = (t >> 4) & 31;
    int blk   = t >> 9;
    int ntile = blk % NT1;
    int kstep = blk / NT1;
    int N     = ntile * 16 + (lane & 15);
    int K     = kstep * 32 + (j & 8) * 2 + (lane >> 4) * 8 + (j & 7);
    w1swz[t] = (_Float16)W1[(size_t)K * D_HID + N];
}

// W2 (f32 [256,40]) -> padded B-fragment blocks [KS2][NT2], zeros for N>=40
__global__ void k_swz_w2(const float* __restrict__ W2, _Float16* __restrict__ w2swz) {
    int t = blockIdx.x * blockDim.x + threadIdx.x;   // 256*48 halves
    if (t >= D_HID * D_OUT_PAD) return;
    int j     = t & 15;
    int lane  = (t >> 4) & 31;
    int blk   = t >> 9;
    int ntile = blk % NT2;
    int kstep = blk / NT2;
    int N     = ntile * 16 + (lane & 15);
    int K     = kstep * 32 + (j & 8) * 2 + (lane >> 4) * 8 + (j & 7);
    w2swz[t] = (N < D_OUT) ? (_Float16)W2[(size_t)K * D_OUT + N] : (_Float16)0.0f;
}

// ---------------- GEMM1: h1[N,256] = xswz @ w1swz ---------------------------
// Block = 8 waves, one column-group g (128 cols), 8 consecutive row tiles.
// W1 fragments for this g (16 ksteps x 8 ntiles x 1 KB = 128 KB) are staged
// into LDS once per block via async global->LDS copies; A is double-buffered.
__global__ void __launch_bounds__(256) k_gemm1(const _Float16* __restrict__ xswz,
                                               const _Float16* __restrict__ w1swz,
                                               float* __restrict__ h1) {
    __shared__ _Float16 bsm[KS1 * 8 * 32 * 16];      // 65536 halves = 128 KB
    const int tid  = threadIdx.x;
    const int lane = tid & 31;
    const int g    = blockIdx.x & 1;                  // column half
    const int m    = (blockIdx.x >> 1) * 8 + (tid >> 5);

    // LDS base byte address of bsm (escapes bsm into the asm -> keeps aliasing)
    const unsigned lbase = (unsigned)(size_t)(void*)bsm;

    // stage: 8192 x 16B chunks; 32 per thread (ASYNCcnt max 63, ok)
    for (int i = tid; i < KS1 * 8 * 32 * 2; i += 256) {
        int b2  = i >> 6;                             // dest fragment block
        int rem = i & 63;                             // 16B chunk in block
        int ks  = b2 >> 3;
        int nt  = b2 & 7;
        const _Float16* gp = w1swz + ((size_t)(ks * NT1 + g * 8 + nt) * 512 + rem * 8);
        async_g2l_b128(lbase + (unsigned)(i * 16), gp);
    }
    wait_asynccnt0();
    __syncthreads();

    if (m < MT) {
        v8f acc[8];
#pragma unroll
        for (int nt = 0; nt < 8; ++nt) acc[nt] = (v8f){0.f,0.f,0.f,0.f,0.f,0.f,0.f,0.f};

        v16h a = *(const v16h*)(xswz + (((size_t)m * KS1) * 32 + lane) * 16);
        for (int ks = 0; ks < KS1; ++ks) {
            v16h an = a;
            if (ks + 1 < KS1)   // prefetch next A during the WMMA burst
                an = *(const v16h*)(xswz + (((size_t)m * KS1 + ks + 1) * 32 + lane) * 16);
#pragma unroll
            for (int nt = 0; nt < 8; ++nt) {
                v16h b = *(const v16h*)(bsm + ((ks * 8 + nt) * 32 + lane) * 16);
                acc[nt] = __builtin_amdgcn_wmma_f32_16x16x32_f16(
                    false, a, false, b, (short)0, acc[nt], false, false);
            }
            a = an;
        }
        int rbase = m * 16 + (lane >> 4) * 8;
        int cbase = g * 128 + (lane & 15);
#pragma unroll
        for (int nt = 0; nt < 8; ++nt)
#pragma unroll
            for (int r = 0; r < 8; ++r)
                h1[(size_t)(rbase + r) * D_HID + cbase + nt * 16] = acc[nt][r];
    }
}

// ---------------- edge scatter, layer 1 (4 edges x 64 lanes per block) ------
__global__ void k_scatter1(const int* __restrict__ src, const int* __restrict__ dst,
                           const float* __restrict__ dinv,
                           const float* __restrict__ h1, float* __restrict__ agg1) {
    int e = blockIdx.x * 4 + (threadIdx.x >> 6);
    int q = threadIdx.x & 63;                         // float4 lane
    int s = src[e], d = dst[e];
    float norm = dinv[s] * dinv[d];
    float4 v = ((const float4*)(h1 + (size_t)s * D_HID))[q];
    float* ap = agg1 + (size_t)d * D_HID + q * 4;
    atomicAdd(ap + 0, v.x * norm);
    atomicAdd(ap + 1, v.y * norm);
    atomicAdd(ap + 2, v.z * norm);
    atomicAdd(ap + 3, v.w * norm);
}

// ------- finalize1: + self-loop + bias, ReLU, f32->f16 swizzled A store -----
__global__ void k_finalize1(const float* __restrict__ agg1, const float* __restrict__ h1,
                            const float* __restrict__ dinv, const float* __restrict__ b1,
                            _Float16* __restrict__ h2swz) {
    int i = blockIdx.x;
    int c = threadIdx.x;                  // 0..255
    float di = dinv[i];
    float v = agg1[(size_t)i * D_HID + c] + h1[(size_t)i * D_HID + c] * di * di + b1[c];
    v = fmaxf(v, 0.0f);
    // inverse A-fragment mapping for (row=i, K=c), KS2 ksteps
    int kstep = c >> 5;
    int kl    = c & 31;
    int grp   = kl >> 4;
    int rem   = kl & 15;
    int half  = rem >> 3;
    int rr    = rem & 7;
    int j     = (grp * 4 + (rr >> 1)) * 2 + (rr & 1);
    int lane  = half * 16 + (i & 15);
    size_t idx = (((size_t)(i >> 4) * KS2 + kstep) * 32 + lane) * 16 + j;
    h2swz[idx] = (_Float16)v;
}

// ---------------- GEMM2: lin2[N,48] = h2swz @ w2swz (16x48 per wave) --------
// Entire padded W2 fragment slab (24 KB) staged into LDS via async copies.
__global__ void __launch_bounds__(256) k_gemm2(const _Float16* __restrict__ h2swz,
                                               const _Float16* __restrict__ w2swz,
                                               float* __restrict__ lin2) {
    __shared__ _Float16 bsm[KS2 * NT2 * 32 * 16];     // 12288 halves = 24 KB
    const int tid  = threadIdx.x;
    const int lane = tid & 31;
    const int m    = blockIdx.x * 8 + (tid >> 5);

    const unsigned lbase = (unsigned)(size_t)(void*)bsm;   // escape bsm

    for (int i = tid; i < KS2 * NT2 * 32 * 2; i += 256)   // 1536 x 16B chunks
        async_g2l_b128(lbase + (unsigned)(i * 16), w2swz + (size_t)i * 8);
    wait_asynccnt0();
    __syncthreads();

    if (m < MT) {
        v8f acc[NT2];
#pragma unroll
        for (int nt = 0; nt < NT2; ++nt) acc[nt] = (v8f){0.f,0.f,0.f,0.f,0.f,0.f,0.f,0.f};

        v16h a = *(const v16h*)(h2swz + (((size_t)m * KS2) * 32 + lane) * 16);
        for (int ks = 0; ks < KS2; ++ks) {
            v16h an = a;
            if (ks + 1 < KS2)
                an = *(const v16h*)(h2swz + (((size_t)m * KS2 + ks + 1) * 32 + lane) * 16);
#pragma unroll
            for (int nt = 0; nt < NT2; ++nt) {
                v16h b = *(const v16h*)(bsm + ((ks * NT2 + nt) * 32 + lane) * 16);
                acc[nt] = __builtin_amdgcn_wmma_f32_16x16x32_f16(
                    false, a, false, b, (short)0, acc[nt], false, false);
            }
            a = an;
        }
        int rbase = m * 16 + (lane >> 4) * 8;
        int cbase = lane & 15;
#pragma unroll
        for (int nt = 0; nt < NT2; ++nt)
#pragma unroll
            for (int r = 0; r < 8; ++r)
                lin2[(size_t)(rbase + r) * D_OUT_PAD + cbase + nt * 16] = acc[nt][r];
    }
}

// ---------------- edge scatter, layer 2 (8 edges x 40 ch per block) ---------
__global__ void k_scatter2(const int* __restrict__ src, const int* __restrict__ dst,
                           const float* __restrict__ dinv,
                           const float* __restrict__ lin2, float* __restrict__ agg2) {
    int le = threadIdx.x / D_OUT;
    int c  = threadIdx.x % D_OUT;
    int e  = blockIdx.x * 8 + le;
    if (e >= E_EDGES) return;
    int s = src[e], d = dst[e];
    float norm = dinv[s] * dinv[d];
    atomicAdd(&agg2[(size_t)d * D_OUT + c], lin2[(size_t)s * D_OUT_PAD + c] * norm);
}

// ------- finalize2: + self-loop + bias, log_softmax (1 wave32 / node) -------
__global__ void k_finalize2(const float* __restrict__ agg2, const float* __restrict__ lin2,
                            const float* __restrict__ dinv, const float* __restrict__ b2,
                            float* __restrict__ out) {
    int i    = blockIdx.x * 8 + (threadIdx.x >> 5);
    int lane = threadIdx.x & 31;
    if (i >= N_NODES) return;
    float di2 = dinv[i] * dinv[i];

    float v0 = agg2[(size_t)i * D_OUT + lane] +
               lin2[(size_t)i * D_OUT_PAD + lane] * di2 + b2[lane];
    float v1 = -INFINITY;
    if (lane < D_OUT - 32)
        v1 = agg2[(size_t)i * D_OUT + lane + 32] +
             lin2[(size_t)i * D_OUT_PAD + lane + 32] * di2 + b2[lane + 32];

    float mx = fmaxf(v0, v1);
#pragma unroll
    for (int off = 16; off > 0; off >>= 1)
        mx = fmaxf(mx, __shfl_xor(mx, off, 32));

    float s = __expf(v0 - mx) + ((lane < D_OUT - 32) ? __expf(v1 - mx) : 0.0f);
#pragma unroll
    for (int off = 16; off > 0; off >>= 1)
        s += __shfl_xor(s, off, 32);

    float lse = __logf(s);
    out[(size_t)i * D_OUT + lane] = v0 - mx - lse;
    if (lane < D_OUT - 32)
        out[(size_t)i * D_OUT + lane + 32] = v1 - mx - lse;
}

// ---------------------------------------------------------------------------
extern "C" void kernel_launch(void* const* d_in, const int* in_sizes, int n_in,
                              void* d_out, int out_size, void* d_ws, size_t ws_size,
                              hipStream_t stream) {
    const float* x    = (const float*)d_in[0];
    const int*   ei   = (const int*)d_in[1];          // int32 (JAX default x64-off)
    const float* W1   = (const float*)d_in[2];
    const float* b1   = (const float*)d_in[3];
    const float* W2   = (const float*)d_in[4];
    const float* b2   = (const float*)d_in[5];
    const int*   src  = ei;
    const int*   dst  = ei + E_EDGES;
    float* out        = (float*)d_out;

    // Workspace layout (regions A/B time-aliased; all offsets 256 B aligned)
    char* ws = (char*)d_ws;
    const size_t SZ_A   = (size_t)N_NODES * D_IN  * 2;       // 102,400,000
    const size_t SZ_B   = (size_t)N_NODES * D_HID * 4;       // 102,400,000
    const size_t SZ_C   = (size_t)N_NODES * D_HID * 2;       //  51,200,000
    _Float16* xswz  = (_Float16*)(ws + 0);                   // region A (phase 1)
    float*    agg1  = (float*)(ws + 0);                      // region A (phase 2)
    float*    agg2  = (float*)(ws + 0);                      // region A (phase 3)
    float*    h1    = (float*)(ws + SZ_A);                   // region B (phase 1-2)
    float*    lin2  = (float*)(ws + SZ_A);                   // region B (phase 3)
    _Float16* h2swz = (_Float16*)(ws + SZ_A + SZ_B);         // region C
    _Float16* w1swz = (_Float16*)(ws + SZ_A + SZ_B + SZ_C);
    _Float16* w2swz = (_Float16*)(ws + SZ_A + SZ_B + SZ_C + (size_t)D_IN * D_HID * 2);
    float*    dinv  = (float*)(ws + SZ_A + SZ_B + SZ_C + (size_t)D_IN * D_HID * 2
                                                      + (size_t)D_HID * D_OUT_PAD * 2);

    // 1) degrees -> dinv (self-loops included via init=1)
    k_deg_init <<<(N_NODES + 255) / 256, 256, 0, stream>>>(dinv);
    k_deg_count<<<(E_EDGES + 255) / 256, 256, 0, stream>>>(dst, dinv);
    k_dinv     <<<(N_NODES + 255) / 256, 256, 0, stream>>>(dinv);

    // 2) fragment swizzles
    k_swz_x <<<(N_NODES * D_IN + 255) / 256, 256, 0, stream>>>(x, xswz);
    k_swz_w1<<<(D_IN * D_HID + 255) / 256, 256, 0, stream>>>(W1, w1swz);
    k_swz_w2<<<(D_HID * D_OUT_PAD + 255) / 256, 256, 0, stream>>>(W2, w2swz);

    // 3) GEMM1 (WMMA, async-LDS staged B): h1 = x @ W1
    k_gemm1<<<((MT + 7) / 8) * 2, 256, 0, stream>>>(xswz, w1swz, h1);

    // 4) layer-1 aggregation (region A now free -> agg1)
    k_zero_f4 <<<((N_NODES * D_HID / 4) + 255) / 256, 256, 0, stream>>>((float4*)agg1,
                                                                        N_NODES * D_HID / 4);
    k_scatter1<<<E_EDGES / 4, 256, 0, stream>>>(src, dst, dinv, h1, agg1);
    k_finalize1<<<N_NODES, 256, 0, stream>>>(agg1, h1, dinv, b1, h2swz);

    // 5) GEMM2 (WMMA, async-LDS staged B): lin2 = relu(h) @ W2
    k_gemm2<<<(MT + 7) / 8, 256, 0, stream>>>(h2swz, w2swz, lin2);

    // 6) layer-2 aggregation (region A reused -> agg2)
    k_zero_f4 <<<((N_NODES * D_OUT / 4) + 255) / 256, 256, 0, stream>>>((float4*)agg2,
                                                                        N_NODES * D_OUT / 4);
    k_scatter2<<<E_EDGES / 8, 320, 0, stream>>>(src, dst, dinv, lin2, agg2);

    // 7) bias + log_softmax -> d_out
    k_finalize2<<<(N_NODES + 7) / 8, 256, 0, stream>>>(agg2, lin2, dinv, b2, out);
}
